// Rcnn_24395414241614
// MI455X (gfx1250) — matrix operand order repaired and checked by reading, not compiled, not run on hardware
//
#include <hip/hip_runtime.h>
#include <hip/hip_bf16.h>

// ---------------------------------------------------------------------------
// R-CNN box head for MI455X (gfx1250, wave32, WMMA + async global->LDS).
// Pipeline: transpose/convert weights to f16 -> ROI align to f16 activations
// -> 3 chained WMMA f32<-f16 GEMMs (LDS double-buffered via
// global_load_async_to_lds_b128 / s_wait_asynccnt, pairwise-unrolled so the
// ping-pong buffer index is compile-time) -> softmax / box decode.
// ---------------------------------------------------------------------------

typedef __attribute__((ext_vector_type(16))) _Float16 v16h;
typedef __attribute__((ext_vector_type(8)))  float    v8f;

#define NROI   1000
#define MPAD   1024
#define KFC6   12544      // 256*7*7
#define NH     1024
#define NHEAD  256        // 87 real columns padded to block tile

// LDS tile row pitch (halves): 32 data + 8 pad -> 80B rows, 16B aligned,
// spreads fragment reads across banks.
#define LDSPITCH 40

// workspace layout (bytes)
#define OFF_X0H   ((size_t)0)                          // [1024][12544] f16
#define OFF_WT6   (OFF_X0H + (size_t)MPAD*KFC6*2)      // [1024][12544] f16 (fc6_w^T)
#define OFF_WT7   (OFF_WT6 + (size_t)NH*KFC6*2)        // [1024][1024]  f16 (fc7_w^T)
#define OFF_WTH   (OFF_WT7 + (size_t)NH*NH*2)          // [256][1024]   f16 (head w^T)
#define OFF_H1H   (OFF_WTH + (size_t)NHEAD*NH*2)       // [1024][1024]  f16
#define OFF_H2H   (OFF_H1H + (size_t)MPAD*NH*2)        // [1024][1024]  f16
#define OFF_HO    (OFF_H2H + (size_t)MPAD*NH*2)        // [1024][256]   f32

// ---------------------------------------------------------------------------
// CDNA5 async global->LDS copy (16B per lane), tracked by ASYNCcnt.
// ---------------------------------------------------------------------------
__device__ __forceinline__ void async_copy_b128(unsigned lds_byte_addr,
                                                const void* gaddr) {
  asm volatile("global_load_async_to_lds_b128 %0, %1, off"
               :: "v"(lds_byte_addr), "v"(gaddr)
               : "memory");
}
#define WAIT_ASYNCCNT(n) asm volatile("s_wait_asynccnt %0" :: "n"(n) : "memory")

// flat LDS pointer -> 32-bit LDS byte address (addr[31:0] is the LDS offset)
__device__ __forceinline__ unsigned lds_addr32(const void* p) {
  return (unsigned)(uintptr_t)p;
}

// ---------------------------------------------------------------------------
// Tiled transpose + fp32->f16 convert:  out[n][k] = (f16) in[k][n]
// ---------------------------------------------------------------------------
__global__ __launch_bounds__(256) void transpose_f32_to_f16(
    const float* __restrict__ in, _Float16* __restrict__ out, int K, int N) {
  __shared__ float tile[32][33];
  const int tx = threadIdx.x, ty = threadIdx.y;       // (32, 8)
  const int kb = blockIdx.y * 32, nb = blockIdx.x * 32;
#pragma unroll
  for (int i = 0; i < 4; ++i) {
    int k = kb + ty + i * 8, n = nb + tx;
    if (k < K && n < N) tile[ty + i * 8][tx] = in[(size_t)k * N + n];
  }
  __syncthreads();
#pragma unroll
  for (int i = 0; i < 4; ++i) {
    int n = nb + ty + i * 8, k = kb + tx;
    if (n < N && k < K) out[(size_t)n * K + k] = (_Float16)tile[tx][ty + i * 8];
  }
}

// head weights: columns 0..80 = cls, 81..84 = box, 85 = cos, 86 = sin
__global__ void pack_head_weights(const float* __restrict__ cls_w,
                                  const float* __restrict__ box_w,
                                  const float* __restrict__ cos_w,
                                  const float* __restrict__ sin_w,
                                  _Float16* __restrict__ wth /* [256][1024] */) {
  int idx = blockIdx.x * blockDim.x + threadIdx.x;
  if (idx >= 87 * 1024) return;
  int n = idx >> 10;          // output column of the head GEMM
  int k = idx & 1023;
  float v;
  if (n < 81)      v = cls_w[(size_t)k * 81 + n];
  else if (n < 85) v = box_w[(size_t)k * 4 + (n - 81)];
  else if (n == 85) v = cos_w[k];
  else              v = sin_w[k];
  wth[(size_t)n * 1024 + k] = (_Float16)v;
}

// ---------------------------------------------------------------------------
// ROI align (aligned=True, sampling_ratio=1, edge-padded feature, crop 14)
// + 2x2 avg pool -> f16 activation row [c*49 + oy*7 + ox].
// One block per ROI, one thread per channel.
// ---------------------------------------------------------------------------
__global__ __launch_bounds__(256) void roi_pool_kernel(
    const float* __restrict__ proposals,
    const float* __restrict__ p2, const float* __restrict__ p3,
    const float* __restrict__ p4, const float* __restrict__ p5,
    _Float16* __restrict__ x0h) {
  const int n = blockIdx.x;
  const int c = threadIdx.x;

  const float px1 = proposals[n * 4 + 0], py1 = proposals[n * 4 + 1];
  const float px2 = proposals[n * 4 + 2], py2 = proposals[n * 4 + 3];
  const float bw = px2 - px1, bh = py2 - py1;
  float lv = 4.0f + log2f(sqrtf(bh * bw) / 224.0f + 1e-6f);
  int lvl = (int)lv;                      // trunc-toward-zero like astype(int32)
  lvl = min(max(lvl, 2), 5);
  const int li = lvl - 2;

  const float* feat; int H, W; float s;
  if (li == 0)      { feat = p2; H = 200; W = 336; s = 0.25f;    }
  else if (li == 1) { feat = p3; H = 100; W = 168; s = 0.125f;   }
  else if (li == 2) { feat = p4; H = 50;  W = 84;  s = 0.0625f;  }
  else              { feat = p5; H = 25;  W = 42;  s = 0.03125f; }

  const float bx1 = px1 * s + 1.0f, by1 = py1 * s + 1.0f;
  const float bx2 = px2 * s + 1.0f, by2 = py2 * s + 1.0f;
  const float sizeXf = (float)(W + 2), sizeYf = (float)(H + 2);
  const float* fc = feat + (size_t)c * H * W;
  _Float16* out = x0h + (size_t)n * KFC6 + c * 49;

  for (int oy = 0; oy < 7; ++oy) {
    float racc[7];
#pragma unroll
    for (int t = 0; t < 7; ++t) racc[t] = 0.0f;
    for (int dy = 0; dy < 2; ++dy) {
      const int sy = oy * 2 + dy;
      const float ys = by1 + (sy + 0.5f) * (1.0f / 14.0f) * (by2 - by1) - 0.5f;
      const bool vy = (ys >= -1.0f) && (ys <= sizeYf);
      const float yc = fminf(fmaxf(ys, 0.0f), sizeYf - 1.0f);
      const int y0 = (int)floorf(yc);
      const int y1i = min(y0 + 1, H + 1);             // size-1 in padded coords
      const float fy = yc - (float)y0;
      const int ym0 = min(max(y0 - 1, 0), H - 1);
      const int ym1 = min(max(y1i - 1, 0), H - 1);
      const float* row0 = fc + (size_t)ym0 * W;
      const float* row1 = fc + (size_t)ym1 * W;
      const float wy0 = 1.0f - fy, wy1 = fy;
      for (int sx = 0; sx < 14; ++sx) {
        const float xs = bx1 + (sx + 0.5f) * (1.0f / 14.0f) * (bx2 - bx1) - 0.5f;
        const bool vx = (xs >= -1.0f) && (xs <= sizeXf);
        const float xc = fminf(fmaxf(xs, 0.0f), sizeXf - 1.0f);
        const int x0 = (int)floorf(xc);
        const int x1i = min(x0 + 1, W + 1);
        const float fx = xc - (float)x0;
        const int xm0 = min(max(x0 - 1, 0), W - 1);
        const int xm1 = min(max(x1i - 1, 0), W - 1);
        const float v = (row0[xm0] * (1.0f - fx) + row0[xm1] * fx) * wy0 +
                        (row1[xm0] * (1.0f - fx) + row1[xm1] * fx) * wy1;
        if (vx && vy) racc[sx >> 1] += v;
      }
    }
#pragma unroll
    for (int ox = 0; ox < 7; ++ox) out[oy * 7 + ox] = (_Float16)(racc[ox] * 0.25f);
  }
}

// ---------------------------------------------------------------------------
// WMMA fragment loaders from LDS tiles (row pitch LDSPITCH halves).
// A 16x32 frag: lane r=l&15 (row), kg=l>>4; halves 0..7 -> K=kg*8+0..7,
//               halves 8..15 -> K=16+kg*8+0..7  (two ds_load_b128).
// B 32x16 frag: lane c=l&15 (col), kg=l>>4; halves 0..7 -> K=kg*8+0..7,
//               halves 8..15 -> K=16+kg*8+0..7  (two ds_load_b128).
// ---------------------------------------------------------------------------
__device__ __forceinline__ v16h lds_a_frag(const _Float16* __restrict__ sA,
                                           int row0, int lane) {
  const int r = lane & 15, kg = lane >> 4;
  const _Float16* p = sA + (row0 + r) * LDSPITCH + kg * 8;
  union { v16h h; uint4 q[2]; } u;
  u.q[0] = *(const uint4*)(p);
  u.q[1] = *(const uint4*)(p + 16);
  return u.h;
}

__device__ __forceinline__ v16h lds_b_frag(const _Float16* __restrict__ sB,
                                           int col0, int lane) {
  const int c = lane & 15, kg = lane >> 4;
  const _Float16* p = sB + (col0 + c) * LDSPITCH + kg * 8;
  union { v16h h; uint4 q[2]; } u;
  u.q[0] = *(const uint4*)(p);
  u.q[1] = *(const uint4*)(p + 16);
  return u.h;
}

// ---------------------------------------------------------------------------
// GEMM: C[M][N] = A[M][K] * B[K][N] (+bias)(+relu), A/Bt f16 in HBM/L2,
// staged through LDS with double-buffered async copies; accum f32.
// 8 waves/block; wave tile 64x64 (16 accum frags); block tile 128x256;
// K-step 32 per stage; #stages (K/32) must be EVEN (392/32/32 here).
// ---------------------------------------------------------------------------
__global__ __launch_bounds__(256) void wmma_gemm_kernel(
    const _Float16* __restrict__ A, const _Float16* __restrict__ Bt,
    const float* __restrict__ bias, _Float16* __restrict__ outH,
    float* __restrict__ outF, int M, int N, int K, int relu) {
  __shared__ _Float16 sA[2][128 * LDSPITCH];
  __shared__ _Float16 sB[2][256 * LDSPITCH];

  const int tid  = threadIdx.x;
  const int lane = tid & 31;
  const int wave = tid >> 5;
  const int wm = wave & 1, wn = wave >> 1;
  const int mBase = blockIdx.y * 128;
  const int nBase = blockIdx.x * 256;
  const int m0 = wm * 64;   // within block tile
  const int n0 = wn * 64;

  // --- per-thread async-copy assignments (16B chunks of 8 halves) ----------
  // A tile: 128 rows x 32 halves = 512 chunks -> 2/thread: row tid>>1,
  //         chunks (tid&1)*2 and +1 (contiguous 16 halves).
  // B tile: 256 rows x 32 halves = 1024 chunks -> 4/thread: row tid.
  const int arow = tid >> 1, ac0 = (tid & 1) * 2;
  const _Float16* gA = A  + (size_t)(mBase + arow) * K + ac0 * 8;
  const _Float16* gB = Bt + (size_t)(nBase + tid) * K;
  unsigned lA[2][2], lB[2][4];
#pragma unroll
  for (int b = 0; b < 2; ++b) {
    lA[b][0] = lds_addr32(&sA[b][arow * LDSPITCH + ac0 * 8]);
    lA[b][1] = lds_addr32(&sA[b][arow * LDSPITCH + ac0 * 8 + 8]);
#pragma unroll
    for (int j = 0; j < 4; ++j)
      lB[b][j] = lds_addr32(&sB[b][tid * LDSPITCH + j * 8]);
  }

  v8f acc[4][4];
#pragma unroll
  for (int i = 0; i < 4; ++i)
#pragma unroll
    for (int j = 0; j < 4; ++j) {
      v8f z = {0.f, 0.f, 0.f, 0.f, 0.f, 0.f, 0.f, 0.f};
      acc[i][j] = z;
    }

  // issue the 6 async copies for one K-stage into buffer b (b is a literal
  // at every call site so the LDS addresses are compile-time selected)
  auto issue = [&](int b, int kb) {
    async_copy_b128(lA[b][0], gA + kb);
    async_copy_b128(lA[b][1], gA + kb + 8);
    async_copy_b128(lB[b][0], gB + kb);
    async_copy_b128(lB[b][1], gB + kb + 8);
    async_copy_b128(lB[b][2], gB + kb + 16);
    async_copy_b128(lB[b][3], gB + kb + 24);
  };

  auto compute = [&](int b) {
    v16h a[4];
#pragma unroll
    for (int i = 0; i < 4; ++i) a[i] = lds_a_frag(sA[b], m0 + 16 * i, lane);
#pragma unroll
    for (int j = 0; j < 4; ++j) {
      v16h bb = lds_b_frag(sB[b], n0 + 16 * j, lane);
#pragma unroll
      for (int i = 0; i < 4; ++i)
        acc[i][j] = __builtin_amdgcn_wmma_f32_16x16x32_f16(
            false, a[i], false, bb, (short)0, acc[i][j], false, false);
    }
  };

  const int nk = K >> 5;          // K / 32 stages (even)
  issue(0, 0);
  for (int t = 0; t < nk; t += 2) {
    // first half of pair: stage t in buf0
    issue(1, (t + 1) << 5);
    WAIT_ASYNCCNT(6);             // stage t retired (async loads in order)
    __syncthreads();
    compute(0);
    __syncthreads();              // all waves done reading buf0
    // second half of pair: stage t+1 in buf1
    if (t + 2 < nk) {
      issue(0, (t + 2) << 5);
      WAIT_ASYNCCNT(6);           // stage t+1 retired
    } else {
      WAIT_ASYNCCNT(0);
    }
    __syncthreads();
    compute(1);
    __syncthreads();              // all waves done reading buf1
  }

  // C/D layout: row = (lane>>4)*8 + vgpr, col = lane&15 within each 16x16 frag
  const int cn = lane & 15, rm = (lane >> 4) * 8;
#pragma unroll
  for (int i = 0; i < 4; ++i)
#pragma unroll
    for (int j = 0; j < 4; ++j)
#pragma unroll
      for (int v = 0; v < 8; ++v) {
        const int row = mBase + m0 + i * 16 + rm + v;
        const int col = nBase + n0 + j * 16 + cn;
        float val = acc[i][j][v];
        if (bias) val += bias[col];
        if (relu) val = fmaxf(val, 0.0f);
        if (outH) outH[(size_t)row * N + col] = (_Float16)val;
        else      outF[(size_t)row * N + col] = val;
      }
}

// ---------------------------------------------------------------------------
// Epilogue: bias adds, softmax over 81 classes, box decode, output scatter.
// Output layout (floats): scores[1000*81] | logits[1000*81] | box_logits[4000]
//                         | cos[1000] | sin[1000] | out_boxes[4000]
// ---------------------------------------------------------------------------
__global__ void head_finalize(const float* __restrict__ ho /* [1024][256] */,
                              const float* __restrict__ proposals,
                              const float* __restrict__ im_shape,
                              const float* __restrict__ cls_b,
                              const float* __restrict__ box_b,
                              const float* __restrict__ cos_b,
                              const float* __restrict__ sin_b,
                              float* __restrict__ out) {
  const int n = blockIdx.x * blockDim.x + threadIdx.x;
  if (n >= NROI) return;
  const float* r = ho + (size_t)n * NHEAD;
  float* sc = out + (size_t)n * 81;
  float* lg = out + 81000 + (size_t)n * 81;

  float mx = -3.4e38f;
  for (int c = 0; c < 81; ++c) {
    const float v = r[c] + cls_b[c];
    lg[c] = v;
    mx = fmaxf(mx, v);
  }
  float sum = 0.0f;
  for (int c = 0; c < 81; ++c) {
    const float e = __expf(lg[c] - mx);
    sc[c] = e;
    sum += e;
  }
  const float inv = 1.0f / sum;
  for (int c = 0; c < 81; ++c) sc[c] *= inv;

  float bl[4];
#pragma unroll
  for (int j = 0; j < 4; ++j) {
    bl[j] = r[81 + j] + box_b[j];
    out[162000 + n * 4 + j] = bl[j];
  }
  out[166000 + n] = r[85] + cos_b[0];
  out[167000 + n] = r[86] + sin_b[0];

  // decode (REG_WEIGHTS = [10,10,5,5]; clip tw/th at log(1344/16)=log(84))
  const float tx = bl[0] * 0.1f, ty = bl[1] * 0.1f;
  const float tw = bl[2] * 0.2f, th = bl[3] * 0.2f;
  const float x1 = proposals[n * 4 + 0], y1 = proposals[n * 4 + 1];
  const float x2 = proposals[n * 4 + 2], y2 = proposals[n * 4 + 3];
  const float wa = x2 - x1, ha = y2 - y1;
  const float xa = (x1 + x2) * 0.5f, ya = (y1 + y2) * 0.5f;
  const float clipv = logf(84.0f);
  const float wb = expf(fminf(tw, clipv)) * wa;
  const float hb = expf(fminf(th, clipv)) * ha;
  const float xb = tx * wa + xa, yb = ty * ha + ya;
  const float b0 = xb - wb * 0.5f, b1 = yb - hb * 0.5f;
  const float b2 = xb + wb * 0.5f, b3 = yb + hb * 0.5f;
  const float s0 = im_shape[0], s1 = im_shape[1];  // reference clips x by H, y by W
  out[168000 + n * 4 + 0] = fminf(fmaxf(b0, 0.0f), s0);
  out[168000 + n * 4 + 1] = fminf(fmaxf(b1, 0.0f), s1);
  out[168000 + n * 4 + 2] = fminf(fmaxf(b2, 0.0f), s0);
  out[168000 + n * 4 + 3] = fminf(fmaxf(b3, 0.0f), s1);
}

// ---------------------------------------------------------------------------
extern "C" void kernel_launch(void* const* d_in, const int* in_sizes, int n_in,
                              void* d_out, int out_size, void* d_ws, size_t ws_size,
                              hipStream_t stream) {
  const float* im_shape  = (const float*)d_in[0];
  const float* proposals = (const float*)d_in[1];
  const float* fc6_w = (const float*)d_in[2];
  const float* fc6_b = (const float*)d_in[3];
  const float* fc7_w = (const float*)d_in[4];
  const float* fc7_b = (const float*)d_in[5];
  const float* cls_w = (const float*)d_in[6];
  const float* cls_b = (const float*)d_in[7];
  const float* box_w = (const float*)d_in[8];
  const float* box_b = (const float*)d_in[9];
  const float* cos_w = (const float*)d_in[10];
  const float* cos_b = (const float*)d_in[11];
  const float* sin_w = (const float*)d_in[12];
  const float* sin_b = (const float*)d_in[13];
  const float* p2 = (const float*)d_in[14];
  const float* p3 = (const float*)d_in[15];
  const float* p4 = (const float*)d_in[16];
  const float* p5 = (const float*)d_in[17];

  char* ws = (char*)d_ws;
  _Float16* x0h = (_Float16*)(ws + OFF_X0H);
  _Float16* wt6 = (_Float16*)(ws + OFF_WT6);
  _Float16* wt7 = (_Float16*)(ws + OFF_WT7);
  _Float16* wth = (_Float16*)(ws + OFF_WTH);
  _Float16* h1h = (_Float16*)(ws + OFF_H1H);
  _Float16* h2h = (_Float16*)(ws + OFF_H2H);
  float*    hof = (float*)(ws + OFF_HO);

  dim3 tb(32, 8);
  // fc6_w: [12544][1024] -> wt6 [1024][12544] f16
  transpose_f32_to_f16<<<dim3(1024 / 32, KFC6 / 32), tb, 0, stream>>>(fc6_w, wt6, KFC6, 1024);
  // fc7_w: [1024][1024] -> wt7
  transpose_f32_to_f16<<<dim3(32, 32), tb, 0, stream>>>(fc7_w, wt7, 1024, 1024);
  pack_head_weights<<<(87 * 1024 + 255) / 256, 256, 0, stream>>>(cls_w, box_w, cos_w, sin_w, wth);

  roi_pool_kernel<<<NROI, 256, 0, stream>>>(proposals, p2, p3, p4, p5, x0h);

  // FC6: [1024 x 12544] x [12544 x 1024] -> relu f16   (nk = 392, even)
  wmma_gemm_kernel<<<dim3(1024 / 256, MPAD / 128), 256, 0, stream>>>(
      x0h, wt6, fc6_b, h1h, nullptr, MPAD, 1024, KFC6, 1);
  // FC7: [1024 x 1024] x [1024 x 1024] -> relu f16     (nk = 32, even)
  wmma_gemm_kernel<<<dim3(1024 / 256, MPAD / 128), 256, 0, stream>>>(
      h1h, wt7, fc7_b, h2h, nullptr, MPAD, 1024, 1024, 1);
  // heads: [1024 x 1024] x [1024 x 256] -> f32 (bias added in finalize)
  wmma_gemm_kernel<<<dim3(NHEAD / 256, MPAD / 128), 256, 0, stream>>>(
      h2h, wth, nullptr, nullptr, hof, MPAD, NHEAD, 1024, 0);

  head_finalize<<<(NROI + 127) / 128, 128, 0, stream>>>(
      hof, proposals, im_shape, cls_b, box_b, cos_b, sin_b, (float*)d_out);
}
